// Downsample_22144851378642
// MI455X (gfx1250) — matrix operand order repaired and checked by reading, not compiled, not run on hardware
//
#include <hip/hip_runtime.h>
#include <hip/hip_bf16.h>

typedef __attribute__((ext_vector_type(16))) _Float16 v16h;
typedef __attribute__((ext_vector_type(8)))  _Float16 v8h;
typedef __attribute__((ext_vector_type(8)))  float    v8f;

#define NB     4
#define CIN    256
#define HH     128
#define WW     128
#define PP     9        // 3x3 taps
#define CO     288      // GROUP * 9
#define HO     126
#define WO     126
#define KDIM   2304     // CIN * PP
#define GROUPS 32
#define EPSVAL 1e-5f

// GEMM tiles
#define BM 96
#define BN 128
#define BK 64

__device__ __forceinline__ void async_wait_zero() {
#if __has_builtin(__builtin_amdgcn_s_wait_asynccnt)
    __builtin_amdgcn_s_wait_asynccnt(0);
#else
    asm volatile("s_wait_asynccnt 0x0" ::: "memory");
#endif
}

// ---------------------------------------------------------------------------
// Kernel 0: repack conv weights OIHW f32 -> f16 A[co][p*CIN+ci]
// ---------------------------------------------------------------------------
__global__ __launch_bounds__(256) void prep_weights_kernel(
    const float* __restrict__ Wt, _Float16* __restrict__ Aw)
{
    const int idx = blockIdx.x * 256 + threadIdx.x;
    if (idx >= CO * KDIM) return;
    const int co  = idx / KDIM;
    const int rem = idx % KDIM;
    const int p   = rem / CIN;
    const int ci  = rem % CIN;
    Aw[idx] = (_Float16)Wt[(size_t)co * KDIM + (size_t)ci * PP + p];
}

// ---------------------------------------------------------------------------
// Kernel 1: implicit-GEMM conv via WMMA f16 (f32 accumulate)
//   Sigma[n][co][y][x] = sum_{ci,ky,kx} x[n][ci][y+ky][x+kx] * w[co][ci][ky][kx]
// Block: 256 threads (8 waves). Tile BM=96 x BN=128, K-chunk 64.
// A tile staged with GLOBAL_LOAD_ASYNC_TO_LDS_B128 (pure byte copy, ASYNCcnt);
// B tile staged manually (f32->f16 convert + transpose to K-contiguous).
// ---------------------------------------------------------------------------
__global__ __launch_bounds__(256) void conv_wmma_kernel(
    const float* __restrict__ X, const _Float16* __restrict__ Aw,
    float* __restrict__ Sigma)
{
    __shared__ _Float16 Alds[BM * BK];   // [m][k], row stride 64 halves (128 B)
    __shared__ _Float16 Blds[BN * BK];   // [x][k], K-contiguous per column

    const int tid  = threadIdx.x;
    const int ny   = blockIdx.x;         // 0 .. 4*126-1
    const int n    = ny / HO;
    const int y    = ny % HO;
    const int co0  = blockIdx.y * BM;

    const int wave = tid >> 5;
    const int lane = tid & 31;
    const int lm   = lane & 15;          // M (A) / N (B,C) index within tile
    const int lh   = lane >> 4;          // lane half selector
    const int nt   = wave;               // this wave's N-subtile

    const int xL = tid & 127;            // B-fill: spatial column 0..127
    const int ch = tid >> 7;             // B-fill: K-half selector

    v8f acc[6] = {};

    for (int p = 0; p < PP; ++p) {
        const int ky = p / 3, kx = p % 3;
        const int yy = y + ky;
        const int xx = xL + kx;          // <= 127+2 = 129: can exceed WW-1; columns x >= WO are not stored
        for (int cc = 0; cc < 4; ++cc) {
            const int ci0 = cc * BK;

            // ---- stage A (weights): async global->LDS b128 copies ----
            for (int i = tid; i < (BM * BK) / 8; i += 256) {
                const int m = i >> 3, seg = i & 7;
                const unsigned long long gaddr = (unsigned long long)(const void*)
                    (Aw + (size_t)(co0 + m) * KDIM + p * CIN + ci0 + seg * 8);
                const unsigned ldsoff =
                    (unsigned)(unsigned long long)(const void*)&Alds[m * BK + seg * 8];
                asm volatile("global_load_async_to_lds_b128 %0, %1, off"
                             :: "v"(ldsoff), "v"(gaddr) : "memory");
            }

            // ---- stage B (im2col, K-contiguous per column) ----
            #pragma unroll
            for (int j = 0; j < 4; ++j) {
                const int c = ch * 32 + j * 8;
                v8h tmp;
                #pragma unroll
                for (int e = 0; e < 8; ++e) {
                    const float v = X[(((size_t)n * CIN + ci0 + c + e) * HH + yy) * WW + xx];
                    tmp[e] = (_Float16)v;
                }
                *(v8h*)&Blds[xL * BK + c] = tmp;
            }
            if (cc < 3) // warm next ci chunk (global_prefetch_b8)
                __builtin_prefetch(&X[(((size_t)n * CIN + ci0 + BK + ch * 32) * HH + yy) * WW + xx], 0, 0);

            async_wait_zero();           // drain ASYNCcnt: A tile resident in LDS
            __syncthreads();

            // ---- compute: 2 K-steps x 6 M-subtiles of v_wmma_f32_16x16x32_f16 ----
            #pragma unroll
            for (int kk = 0; kk < BK; kk += 32) {
                v16h bfrag;
                {
                    // B layout: VGPR0-3 -> K = lh*8 + 0..7, VGPR4-7 -> +16
                    const v8h b0 = *(const v8h*)&Blds[(nt * 16 + lm) * BK + kk + lh * 8];
                    const v8h b1 = *(const v8h*)&Blds[(nt * 16 + lm) * BK + kk + 16 + lh * 8];
                    #pragma unroll
                    for (int e = 0; e < 8; ++e) { bfrag[e] = b0[e]; bfrag[8 + e] = b1[e]; }
                }
                #pragma unroll
                for (int mt = 0; mt < 6; ++mt) {
                    // A layout: VGPR0-3 -> K = lh*8 + 0..7, VGPR4-7 -> +16
                    v16h afrag;
                    const v8h a0 = *(const v8h*)&Alds[(mt * 16 + lm) * BK + kk + lh * 8];
                    const v8h a1 = *(const v8h*)&Alds[(mt * 16 + lm) * BK + kk + 16 + lh * 8];
                    #pragma unroll
                    for (int e = 0; e < 8; ++e) { afrag[e] = a0[e]; afrag[8 + e] = a1[e]; }
                    acc[mt] = __builtin_amdgcn_wmma_f32_16x16x32_f16(
                        false, afrag, false, bfrag, (short)0, acc[mt], false, false);
                }
            }
            __syncthreads();
        }
    }

    // ---- epilogue: C/D layout lane L -> N=lm, VGPR r -> M = lh*8 + r ----
    const int xo = nt * 16 + lm;
    if (xo < WO) {
        #pragma unroll
        for (int mt = 0; mt < 6; ++mt) {
            #pragma unroll
            for (int r = 0; r < 8; ++r) {
                const int co = co0 + mt * 16 + lh * 8 + r;
                Sigma[(((size_t)n * CO + co) * HO + y) * WO + xo] = acc[mt][r];
            }
        }
    }
}

// ---------------------------------------------------------------------------
// Kernel 2: per-channel mean / rstd over (n, h, w)
// ---------------------------------------------------------------------------
__global__ __launch_bounds__(256) void stats_kernel(
    const float* __restrict__ Sigma, float* __restrict__ Mean, float* __restrict__ Rstd)
{
    __shared__ float ssum[256], ssq[256];
    const int co = blockIdx.x;
    float s = 0.f, q = 0.f;
    for (int i = threadIdx.x; i < NB * HO * WO; i += 256) {
        const int nn = i / (HO * WO);
        const int sp = i % (HO * WO);
        const float v = Sigma[((size_t)nn * CO + co) * (HO * WO) + sp];
        s += v; q += v * v;
    }
    ssum[threadIdx.x] = s; ssq[threadIdx.x] = q;
    __syncthreads();
    for (int off = 128; off > 0; off >>= 1) {
        if (threadIdx.x < off) {
            ssum[threadIdx.x] += ssum[threadIdx.x + off];
            ssq[threadIdx.x]  += ssq[threadIdx.x + off];
        }
        __syncthreads();
    }
    if (threadIdx.x == 0) {
        const float inv = 1.f / (float)(NB * HO * WO);
        const float m   = ssum[0] * inv;
        const float var = ssq[0] * inv - m * m;
        Mean[co] = m;
        Rstd[co] = rsqrtf(var + EPSVAL);
    }
}

// ---------------------------------------------------------------------------
// Kernel 3: norm + channel softmax (online, 2 passes) + dynamic 3x3 conv
// One block per (n, y) row; lanes cover x (coalesced channel-strided reads).
// ---------------------------------------------------------------------------
__global__ __launch_bounds__(128) void dynconv_kernel(
    const float* __restrict__ X, const float* __restrict__ Sigma,
    const float* __restrict__ Mean, const float* __restrict__ Rstd,
    const float* __restrict__ Gamma, const float* __restrict__ Beta,
    float* __restrict__ OutS)
{
    const int ny = blockIdx.x;
    const int n = ny / HO, y = ny % HO;
    const int x = threadIdx.x;
    if (x >= WO) return;
    const size_t sbase = ((size_t)n * CO) * (HO * WO) + (size_t)y * WO + x;

    // pass 1: online softmax max & denom
    float m = -3.4e38f, l = 0.f;
    for (int co = 0; co < CO; ++co) {
        const float v = (Sigma[sbase + (size_t)co * (HO * WO)] - Mean[co]) * Rstd[co]
                        * Gamma[co] + Beta[co];
        const float m2 = fmaxf(m, v);
        l = l * __expf(m - m2) + __expf(v - m2);
        m = m2;
    }
    const float linv = 1.f / l;

    // pass 2: weighted 9-tap gather per group of 8 channels
    for (int g = 0; g < GROUPS; ++g) {
        float acc[8] = {0.f, 0.f, 0.f, 0.f, 0.f, 0.f, 0.f, 0.f};
        for (int p = 0; p < PP; ++p) {
            const int co = g * PP + p;
            const float v = (Sigma[sbase + (size_t)co * (HO * WO)] - Mean[co]) * Rstd[co]
                            * Gamma[co] + Beta[co];
            const float wgt = __expf(v - m) * linv;
            const int ky = p / 3, kx = p % 3;
            #pragma unroll
            for (int c8 = 0; c8 < 8; ++c8) {
                const int chn = g * 8 + c8;
                acc[c8] += X[(((size_t)n * CIN + chn) * HH + y + ky) * WW + x + kx] * wgt;
            }
        }
        #pragma unroll
        for (int c8 = 0; c8 < 8; ++c8)
            OutS[(((size_t)n * CIN + g * 8 + c8) * HO + y) * WO + x] = acc[c8];
    }
}

// ---------------------------------------------------------------------------
// Kernel 4: bilinear align-corners resize 126x126 -> 128x128
// ---------------------------------------------------------------------------
__global__ __launch_bounds__(256) void bilinear_kernel(
    const float* __restrict__ OutS, float* __restrict__ Out)
{
    const int id = blockIdx.x * 256 + threadIdx.x;   // 2^24 elements
    const int Xo = id & 127;
    const int Yo = (id >> 7) & 127;
    const int c  = (id >> 14) & 255;
    const int n  = id >> 22;
    const float sc  = 125.0f / 127.0f;
    const float ysf = Yo * sc;
    const float xsf = Xo * sc;
    const int y0 = (int)ysf;
    const int x0 = (int)xsf;
    const float wy = ysf - (float)y0, wx = xsf - (float)x0;
    const int y1 = min(y0 + 1, HO - 1), x1 = min(x0 + 1, WO - 1);
    const float* pl = OutS + ((size_t)n * CIN + c) * (HO * WO);
    const float a = pl[y0 * WO + x0], b = pl[y0 * WO + x1];
    const float d = pl[y1 * WO + x0], e = pl[y1 * WO + x1];
    const float top = a * (1.f - wx) + b * wx;
    const float bot = d * (1.f - wx) + e * wx;
    Out[id] = top * (1.f - wy) + bot * wy;
}

// ---------------------------------------------------------------------------
extern "C" void kernel_launch(void* const* d_in, const int* in_sizes, int n_in,
                              void* d_out, int out_size, void* d_ws, size_t ws_size,
                              hipStream_t stream) {
    (void)in_sizes; (void)n_in; (void)out_size; (void)ws_size;
    const float* X     = (const float*)d_in[0];
    const float* Wt    = (const float*)d_in[1];
    const float* Gamma = (const float*)d_in[2];
    const float* Beta  = (const float*)d_in[3];
    float*       Out   = (float*)d_out;

    // workspace layout (256-byte aligned slabs)
    char* ws = (char*)d_ws;
    const size_t sigmaBytes = ((size_t)NB * CO * HO * WO * 4 + 255) & ~(size_t)255;
    const size_t outsBytes  = ((size_t)NB * CIN * HO * WO * 4 + 255) & ~(size_t)255;
    float*    Sigma = (float*)ws;            ws += sigmaBytes;
    float*    OutS  = (float*)ws;            ws += outsBytes;
    float*    Mean  = (float*)ws;            ws += 512 * 4;
    float*    Rstd  = (float*)ws;            ws += 512 * 4;
    _Float16* Aw    = (_Float16*)ws;         // CO*KDIM halves

    prep_weights_kernel<<<(CO * KDIM + 255) / 256, 256, 0, stream>>>(Wt, Aw);

    dim3 gconv(NB * HO, CO / BM);            // (504, 3)
    conv_wmma_kernel<<<gconv, 256, 0, stream>>>(X, Aw, Sigma);

    stats_kernel<<<CO, 256, 0, stream>>>(Sigma, Mean, Rstd);

    dynconv_kernel<<<NB * HO, 128, 0, stream>>>(X, Sigma, Mean, Rstd, Gamma, Beta, OutS);

    bilinear_kernel<<<((size_t)NB * CIN * HH * WW) / 256, 256, 0, stream>>>(OutS, Out);
}
